// merge_53008486367352
// MI455X (gfx1250) — matrix operand order repaired and checked by reading, not compiled, not run on hardware
//
#include <hip/hip_runtime.h>

// ---------------------------------------------------------------------------
// Types
// ---------------------------------------------------------------------------
typedef __bf16        v16bf __attribute__((ext_vector_type(16)));
typedef float         v8f   __attribute__((ext_vector_type(8)));
typedef unsigned int  u32x4 __attribute__((ext_vector_type(4)));
typedef float         f32x2 __attribute__((ext_vector_type(2)));
typedef unsigned int   u32;
typedef unsigned short u16;

#define NPOS  16384   // 4 * 64 * 64 flattened spatial positions (half res)
#define CIN   512
#define NRP   256     // channel row-pairs

union AFrag { u32x4 q[2]; v16bf v; };
union BFrag { u32x4 q[2]; v16bf v; };

__device__ __forceinline__ u16 f2bf(float f) {
  u32 u = __float_as_uint(f);
  u32 r = (u + 0x7FFFu + ((u >> 16) & 1u)) >> 16;   // round-to-nearest-even
  return (u16)r;
}
__device__ __forceinline__ float bf2f(u16 h) {
  return __uint_as_float(((u32)h) << 16);
}

// Fragment-swizzled dword address for packed activations.
// rp = channel row-pair (0..255), P = position (0..16383).
// Fragment (kstep ks = rp/16, postile pt = P/16) occupies 256 contiguous
// dwords; lane l owns dwords l*8..l*8+7 in this file's contiguous B k order (k = 16*half + s, same as A):
//   VGPR j, lanes 0-15: K = 2j,2j+1 ; lanes 16-31: K = 16+2j,17+2j.
__device__ __forceinline__ size_t swz_addr(int rp, int P) {
  int ks   = rp >> 4;
  int jj   = rp & 15;
  int lane = ((jj >> 3) << 4) + (P & 15);
  int j    = jj & 7;
  return ((size_t)(P >> 4) * 16 + ks) * 256 + lane * 8 + j;
}

// ---------------------------------------------------------------------------
// 1) Weight prep: fp32 (o,c) 512x512 -> bf16 hi/lo row-major, 4 matrices
// ---------------------------------------------------------------------------
__global__ __launch_bounds__(256) void wprep_k(
    const float* __restrict__ w0, const float* __restrict__ w1,
    const float* __restrict__ w2, const float* __restrict__ w3,
    u16* __restrict__ WH, u16* __restrict__ WL)
{
  int i = blockIdx.x * 256 + threadIdx.x;            // 0 .. 262143
  const float* src[4] = {w0, w1, w2, w3};
#pragma unroll
  for (int mtx = 0; mtx < 4; ++mtx) {
    float v = src[mtx][i];
    u16 h = f2bf(v);
    WH[mtx * 262144 + i] = h;
    WL[mtx * 262144 + i] = f2bf(v - bf2f(h));
  }
}

// ---------------------------------------------------------------------------
// 2) Haar DWT: LH/HL/HH fp32 out; LL as swizzled channel-pair-packed bf16
// ---------------------------------------------------------------------------
__global__ __launch_bounds__(256) void dwt_pack_k(
    const float* __restrict__ x,
    float* __restrict__ LH, float* __restrict__ HL, float* __restrict__ HH,
    u32* __restrict__ LLhi, u32* __restrict__ LLlo)
{
  int idx = blockIdx.x * 256 + threadIdx.x;          // 2^22 threads
  int w2 = idx & 63;
  int h2 = (idx >> 6) & 63;
  int cp = (idx >> 12) & 255;
  int b  = idx >> 20;
  int P  = b * 4096 + h2 * 64 + w2;

  u32 pkhi = 0, pklo = 0;
#pragma unroll
  for (int s = 0; s < 2; ++s) {
    int c = cp * 2 + s;
    const float* px = x + (((size_t)(b * 512 + c) * 128 + h2 * 2) * 128 + w2 * 2);
    f32x2 r0 = *(const f32x2*)px;
    f32x2 r1 = *(const f32x2*)(px + 128);
    float a = r0.x, bb = r0.y, cc = r1.x, dd = r1.y;
    float ll = ( a + bb + cc + dd) * 0.5f;
    float lh = (-a + bb - cc + dd) * 0.5f;
    float hl = (-a - bb + cc + dd) * 0.5f;
    float hh = ( a - bb - cc + dd) * 0.5f;
    size_t o = (size_t)(b * 512 + c) * 4096 + h2 * 64 + w2;
    LH[o] = lh; HL[o] = hl; HH[o] = hh;
    u16 hi = f2bf(ll);
    u16 lo = f2bf(ll - bf2f(hi));
    pkhi |= ((u32)hi) << (16 * s);
    pklo |= ((u32)lo) << (16 * s);
  }
  size_t sa = swz_addr(cp, P);
  LLhi[sa] = pkhi;
  LLlo[sa] = pklo;
}

// ---------------------------------------------------------------------------
// 3) GEMM  D(512 x 16384) = W(512x512) x X(512x16384), split-bf16 x3 WMMA.
//    Wave: 16(o) x 64(pos); block: 8 waves -> 128 x 64.
//    X is fragment-swizzled: each B fragment = 2 x global_load_b128 per array.
//    packOut=1: emit D swizzled bf16 hi/lo (feeds q/k/v GEMMs).
//    packOut=0: emit fp32 plain (o, P).
// ---------------------------------------------------------------------------
__global__ __launch_bounds__(256) void gemm_wmma_k(
    const u16* __restrict__ Whi, const u16* __restrict__ Wlo,
    const u32* __restrict__ Xhi, const u32* __restrict__ Xlo,
    const float* __restrict__ bias,
    float* __restrict__ Dplain,
    u32* __restrict__ Yhi, u32* __restrict__ Ylo,
    int packOut)
{
  const int lane = threadIdx.x & 31;
  const int wave = threadIdx.x >> 5;
  const int half = lane >> 4;          // 0: lanes 0-15, 1: lanes 16-31
  const int m    = lane & 15;
  const int o0   = blockIdx.x * 128 + wave * 16;
  const int p0   = blockIdx.y * 64;
  const int pt0  = blockIdx.y * 4;     // first pos-tile of this block

  v8f acc[4] = {v8f{}, v8f{}, v8f{}, v8f{}};

  for (int kc = 0; kc < CIN; kc += 32) {
    // ---- A fragments (16x32 bf16): lane row = o0+m, K = kc + 16*half + s (contiguous, same as B)
    AFrag ahi, alo;
    const int baseA = (o0 + m) * CIN + kc + half * 16;
    ahi.q[0] = *(const u32x4*)(Whi + baseA);
    ahi.q[1] = *(const u32x4*)(Whi + baseA + 8);
    alo.q[0] = *(const u32x4*)(Wlo + baseA);
    alo.q[1] = *(const u32x4*)(Wlo + baseA + 8);

    const int ks = kc >> 5;
#pragma unroll
    for (int t = 0; t < 4; ++t) {
      // ---- B fragment: 1KiB contiguous, lane owns 32B -> 2 x b128 per array
      const size_t fb = ((size_t)(pt0 + t) * 16 + ks) * 256 + lane * 8;
      BFrag bhi, blo;
      bhi.q[0] = *(const u32x4*)(Xhi + fb);
      bhi.q[1] = *(const u32x4*)(Xhi + fb + 4);
      blo.q[0] = *(const u32x4*)(Xlo + fb);
      blo.q[1] = *(const u32x4*)(Xlo + fb + 4);

      acc[t] = __builtin_amdgcn_wmma_f32_16x16x32_bf16(
          false, ahi.v, false, bhi.v, (short)0, acc[t], false, false);
      acc[t] = __builtin_amdgcn_wmma_f32_16x16x32_bf16(
          false, ahi.v, false, blo.v, (short)0, acc[t], false, false);
      acc[t] = __builtin_amdgcn_wmma_f32_16x16x32_bf16(
          false, alo.v, false, bhi.v, (short)0, acc[t], false, false);
    }
  }

  if (bias != nullptr) {
#pragma unroll
    for (int r = 0; r < 8; ++r) {
      float bv = bias[o0 + r + half * 8];
#pragma unroll
      for (int t = 0; t < 4; ++t) acc[t][r] += bv;
    }
  }

  if (!packOut) {
#pragma unroll
    for (int t = 0; t < 4; ++t) {
      const int col = p0 + t * 16 + m;
#pragma unroll
      for (int r = 0; r < 8; ++r)
        Dplain[(o0 + r + half * 8) * NPOS + col] = acc[t][r];
    }
  } else {
    // acc VGPRs (2rp, 2rp+1) are adjacent channels in the same lane -> pack
#pragma unroll
    for (int t = 0; t < 4; ++t) {
      const int col = p0 + t * 16 + m;
#pragma unroll
      for (int rp = 0; rp < 4; ++rp) {
        float v0 = acc[t][2 * rp], v1 = acc[t][2 * rp + 1];
        u16 h0 = f2bf(v0), h1 = f2bf(v1);
        u16 l0 = f2bf(v0 - bf2f(h0)), l1 = f2bf(v1 - bf2f(h1));
        int row = (o0 >> 1) + rp + half * 4;     // output channel row-pair
        size_t sa = swz_addr(row, col);
        Yhi[sa] = ((u32)h1 << 16) | (u32)h0;
        Ylo[sa] = ((u32)l1 << 16) | (u32)l0;
      }
    }
  }
}

// ---------------------------------------------------------------------------
// 4) zero the BN accumulators
// ---------------------------------------------------------------------------
__global__ void bninit_k(float* __restrict__ bn) {
  bn[blockIdx.x * 256 + threadIdx.x] = 0.0f;
}

// ---------------------------------------------------------------------------
// 5) Local 3x3 attention (zero-pad semantics) + per-channel sum/sumsq
//    One 256-thread block covers 256 consecutive positions of one channel.
// ---------------------------------------------------------------------------
__global__ __launch_bounds__(256) void attn_bn_k(
    const float* __restrict__ Q, const float* __restrict__ K,
    const float* __restrict__ V, float* __restrict__ Z,
    float* __restrict__ bn)
{
  int idx = blockIdx.x * 256 + threadIdx.x;          // 2^23 threads
  int P   = idx & (NPOS - 1);
  int o   = idx >> 14;
  int w   = P & 63;
  int h   = (P >> 6) & 63;

  float q = Q[idx];
  float s[9], vv[9];
  float mx = -1e30f;
  int n = 0;
#pragma unroll
  for (int dh = -1; dh <= 1; ++dh) {
#pragma unroll
    for (int dw = -1; dw <= 1; ++dw) {
      int hh = h + dh, ww = w + dw;
      float kv = 0.0f, vvl = 0.0f;
      if (hh >= 0 && hh < 64 && ww >= 0 && ww < 64) {
        int off = idx + dh * 64 + dw;
        kv  = K[off];
        vvl = V[off];
      }
      float sc = q * kv;                 // zero-pad -> sc = 0, still in softmax
      s[n] = sc; vv[n] = vvl;
      mx = fmaxf(mx, sc);
      ++n;
    }
  }
  float den = 0.0f, num = 0.0f;
#pragma unroll
  for (int i = 0; i < 9; ++i) {
    float e = __expf(s[i] - mx);
    den += e;
    num += e * vv[i];
  }
  float z = num / den;
  Z[idx] = z;

  // block-level reduction (whole block shares channel o)
  float s1 = z, s2 = z * z;
#pragma unroll
  for (int off = 16; off > 0; off >>= 1) {
    s1 += __shfl_xor(s1, off, 32);
    s2 += __shfl_xor(s2, off, 32);
  }
  __shared__ float red[16];
  int lane = threadIdx.x & 31, wv = threadIdx.x >> 5;
  if (lane == 0) { red[wv] = s1; red[8 + wv] = s2; }
  __syncthreads();
  if (threadIdx.x == 0) {
    float t1 = 0.0f, t2 = 0.0f;
#pragma unroll
    for (int i = 0; i < 8; ++i) { t1 += red[i]; t2 += red[8 + i]; }
    atomicAdd(&bn[o], t1);
    atomicAdd(&bn[512 + o], t2);
  }
}

// ---------------------------------------------------------------------------
// 6) BN (from sums) + PReLU + inverse Haar DWT
// ---------------------------------------------------------------------------
__global__ __launch_bounds__(256) void finalize_k(
    const float* __restrict__ Z, const float* __restrict__ bn,
    const float* __restrict__ LH, const float* __restrict__ HL,
    const float* __restrict__ HH,
    const float* __restrict__ gamma, const float* __restrict__ beta,
    const float* __restrict__ prelu_a, float* __restrict__ out)
{
  int idx = blockIdx.x * 256 + threadIdx.x;          // 2^23 threads
  int p2 = idx & 4095;
  int c  = (idx >> 12) & 511;
  int b  = idx >> 21;
  int h2 = p2 >> 6, w2 = p2 & 63;
  int P  = b * 4096 + p2;

  float z   = Z[c * NPOS + P];
  float m   = bn[c] * (1.0f / 16384.0f);
  float var = bn[512 + c] * (1.0f / 16384.0f) - m * m;
  float y   = gamma[c] * (z - m) * rsqrtf(var + 1e-5f) + beta[c];
  float ap  = prelu_a[0];
  y = (y >= 0.0f) ? y : ap * y;

  size_t o4 = (size_t)(b * 512 + c) * 4096 + p2;
  float lh = LH[o4], hl = HL[o4], hh = HH[o4];
  float av = (y - lh - hl + hh) * 0.5f;
  float bv = (y + lh - hl - hh) * 0.5f;
  float cv = (y - lh + hl - hh) * 0.5f;
  float dv = (y + lh + hl + hh) * 0.5f;

  float* po = out + (((size_t)(b * 512 + c) * 128 + 2 * h2) * 128 + 2 * w2);
  f32x2 e; e.x = av; e.y = bv;
  f32x2 f; f.x = cv; f.y = dv;
  *(f32x2*)po         = e;
  *(f32x2*)(po + 128) = f;
}

// ---------------------------------------------------------------------------
// Host launcher
// ---------------------------------------------------------------------------
extern "C" void kernel_launch(void* const* d_in, const int* in_sizes, int n_in,
                              void* d_out, int out_size, void* d_ws, size_t ws_size,
                              hipStream_t stream) {
  const float* x    = (const float*)d_in[0];
  const float* w_c1 = (const float*)d_in[1];
  const float* b_c1 = (const float*)d_in[2];
  const float* w_q  = (const float*)d_in[3];
  const float* w_k  = (const float*)d_in[4];
  const float* w_v  = (const float*)d_in[5];
  const float* gmm  = (const float*)d_in[6];
  const float* bta  = (const float*)d_in[7];
  const float* pa   = (const float*)d_in[8];
  float* out = (float*)d_out;

  char* ws = (char*)d_ws;
  size_t off = 0;
  auto carve = [&](size_t bytes) -> char* {
    char* p = ws + off;
    off = (off + bytes + 255) & ~(size_t)255;
    return p;
  };
  const size_t SZ32M = (size_t)4 * 512 * 4096 * 4;   // 32 MiB
  const size_t SZ16M = (size_t)NRP * NPOS * 4;       // 16 MiB

  float* LH   = (float*)carve(SZ32M);
  float* HL   = (float*)carve(SZ32M);
  float* HH   = (float*)carve(SZ32M);
  u32*   LLhi = (u32*)  carve(SZ16M);
  u32*   LLlo = (u32*)  carve(SZ16M);
  u16*   WH   = (u16*)  carve((size_t)4 * 262144 * 2);
  u16*   WL   = (u16*)  carve((size_t)4 * 262144 * 2);
  u32*   Yhi  = (u32*)  carve(SZ16M);
  u32*   Ylo  = (u32*)  carve(SZ16M);
  float* Q    = (float*)carve(SZ32M);
  float* K    = (float*)carve(SZ32M);
  float* V    = (float*)carve(SZ32M);
  float* bn   = (float*)carve(4096);
  float* Z    = (float*)LLhi;                        // reuse: LL packed is dead after GEMM1

  // 1) weights -> bf16 hi/lo
  wprep_k<<<1024, 256, 0, stream>>>(w_c1, w_q, w_k, w_v, WH, WL);
  // 2) DWT + pack LL (swizzled)
  dwt_pack_k<<<16384, 256, 0, stream>>>(x, LH, HL, HH, LLhi, LLlo);
  // 3) y = W_c1 * LL + b  (swizzled bf16 hi/lo output)
  dim3 gg(4, 256);
  gemm_wmma_k<<<gg, 256, 0, stream>>>(WH + 0 * 262144, WL + 0 * 262144,
                                      LLhi, LLlo, b_c1, nullptr, Yhi, Ylo, 1);
  // 4) q, k, v = W_{q,k,v} * y  (fp32 plain output)
  gemm_wmma_k<<<gg, 256, 0, stream>>>(WH + 1 * 262144, WL + 1 * 262144,
                                      Yhi, Ylo, nullptr, Q, nullptr, nullptr, 0);
  gemm_wmma_k<<<gg, 256, 0, stream>>>(WH + 2 * 262144, WL + 2 * 262144,
                                      Yhi, Ylo, nullptr, K, nullptr, nullptr, 0);
  gemm_wmma_k<<<gg, 256, 0, stream>>>(WH + 3 * 262144, WL + 3 * 262144,
                                      Yhi, Ylo, nullptr, V, nullptr, nullptr, 0);
  // 5) attention + BN partial sums
  bninit_k<<<4, 256, 0, stream>>>(bn);
  attn_bn_k<<<32768, 256, 0, stream>>>(Q, K, V, Z, bn);
  // 6) BN + PReLU + IDWT
  finalize_k<<<32768, 256, 0, stream>>>(Z, bn, LH, HL, HH, gmm, bta, pa, out);
}
